// RSSM_58282706207361
// MI455X (gfx1250) — matrix operand order repaired and checked by reading, not compiled, not run on hardware
//
#include <hip/hip_runtime.h>
#include <hip/hip_bf16.h>

typedef __attribute__((ext_vector_type(16))) _Float16 v16h;
typedef __attribute__((ext_vector_type(8)))  _Float16 v8h;
typedef __attribute__((ext_vector_type(8)))  float    v8f;

#define B_   2048
#define L_   64
#define STOCH 30
#define DETER 200
#define OBS_  200
#define HID_  200
#define ACT_  6
#define OUTC 350

// ---- fragment-packed weight table (each fragment: 32 lanes x 16 halves = 512 halves) ----
// Wrz: NT=26 (r zone 208 + z zone 208), KT=9 (x zone 64 + h zone 224)
// Win: NT=13, KT=2 (x zone 64)        -- W_ih rows 400..600
// Whn: NT=13, KT=7 (h zone 224)       -- W_hh rows 400..600
// Pw1: NT=13, KT=7
// Pw2: NT=4 (mu 32 + pre 32), KT=7
// Qw1: NT=13, KT=14 (obs zone 224 + h zone 224)
// Qw2: NT=4, KT=7
#define F_WRZ   0
#define F_WIN   (F_WRZ + 26*9)
#define F_WHN   (F_WIN + 13*2)
#define F_PW1   (F_WHN + 13*7)
#define F_PW2   (F_PW1 + 13*7)
#define F_QW1   (F_PW2 + 4*7)
#define F_QW2   (F_QW1 + 13*14)
#define F_TOTAL (F_QW2 + 4*7)
#define FRAG_BYTES ((size_t)F_TOTAL * 512 * 2)

// ---- packed bias offsets (floats) ----
#define BO_RZ 0       // 416: (b_ih+b_hh)[r|z] zones of 208
#define BO_IN 416     // 208: b_ih[400..600]
#define BO_HN 624     // 208: b_hh[400..600]
#define BO_P1 832     // 224
#define BO_P2 1056    // 64 (mu 32 | pre 32)
#define BO_Q1 1120    // 224
#define BO_Q2 1344    // 64
#define BIAS_TOTAL 1408

// =====================================================================
// Pack kernel: rewrite weights (f32, torch [out,in] layout) into WMMA-B
// fragment order as f16.  B-fragment layout (16-bit, 32xK16xN):
//   lane l holds column n' = nt*16 + (l%16); halves s=0..7 hold
//   k' = kt*32 + (l/16)*8 + s, halves s=8..15 hold k' = kt*32 + 16 + (l/16)*8 + (s-8).
// =====================================================================
__global__ void pack_frags(const float* __restrict__ W_ih, const float* __restrict__ W_hh,
                           const float* __restrict__ pW1,  const float* __restrict__ pW2,
                           const float* __restrict__ qW1,  const float* __restrict__ qW2,
                           _Float16* __restrict__ frags)
{
  int f = blockIdx.x;
  if (f >= F_TOTAL) return;
#pragma unroll 1
  for (int u = 0; u < 8; ++u) {
    int hidx = threadIdx.x * 8 + u;       // 64 threads x 8 = 512 halves
    int lane = hidx >> 4;
    int j    = (hidx & 15) + ((hidx & 8) ? 8 : 0) - (lane >> 4)*8;
    float val = 0.f;
    if (f < F_WIN) {                       // Wrz
      int rel = f - F_WRZ, nt = rel / 9, kt = rel % 9;
      int np = nt*16 + (lane & 15), kp = kt*32 + (lane >> 4)*16 + j;
      int g = np / 208, idx = np % 208;
      if (idx < 200) {
        int row = g*200 + idx;             // r rows 0..200, z rows 200..400
        if (kp < 64) { if (kp < 36) val = W_ih[row*36 + kp]; }
        else { int k = kp - 64; if (k < 200) val = W_hh[row*200 + k]; }
      }
    } else if (f < F_WHN) {                // Win
      int rel = f - F_WIN, nt = rel / 2, kt = rel % 2;
      int np = nt*16 + (lane & 15), kp = kt*32 + (lane >> 4)*16 + j;
      if (np < 200 && kp < 36) val = W_ih[(400 + np)*36 + kp];
    } else if (f < F_PW1) {                // Whn
      int rel = f - F_WHN, nt = rel / 7, kt = rel % 7;
      int np = nt*16 + (lane & 15), kp = kt*32 + (lane >> 4)*16 + j;
      if (np < 200 && kp < 200) val = W_hh[(400 + np)*200 + kp];
    } else if (f < F_PW2) {                // Pw1
      int rel = f - F_PW1, nt = rel / 7, kt = rel % 7;
      int np = nt*16 + (lane & 15), kp = kt*32 + (lane >> 4)*16 + j;
      if (np < 200 && kp < 200) val = pW1[np*200 + kp];
    } else if (f < F_QW1) {                // Pw2 (mu zone 32 | pre zone 32)
      int rel = f - F_PW2, nt = rel / 7, kt = rel % 7;
      int np = nt*16 + (lane & 15), kp = kt*32 + (lane >> 4)*16 + j;
      int g = np / 32, idx = np % 32;
      if (idx < 30 && kp < 200) val = pW2[(g*30 + idx)*200 + kp];
    } else if (f < F_QW2) {                // Qw1 (K: obs 224 | h 224)
      int rel = f - F_QW1, nt = rel / 14, kt = rel % 14;
      int np = nt*16 + (lane & 15), kp = kt*32 + (lane >> 4)*16 + j;
      if (np < 200) {
        if (kp < 224) { if (kp < 200) val = qW1[np*400 + kp]; }
        else { int k = kp - 224; if (k < 200) val = qW1[np*400 + 200 + k]; }
      }
    } else {                               // Qw2
      int rel = f - F_QW2, nt = rel / 7, kt = rel % 7;
      int np = nt*16 + (lane & 15), kp = kt*32 + (lane >> 4)*16 + j;
      int g = np / 32, idx = np % 32;
      if (idx < 30 && kp < 200) val = qW2[(g*30 + idx)*200 + kp];
    }
    frags[(size_t)f * 512 + hidx] = (_Float16)val;
  }
}

__global__ void pack_bias(const float* __restrict__ b_ih, const float* __restrict__ b_hh,
                          const float* __restrict__ pb1,  const float* __restrict__ pb2,
                          const float* __restrict__ qb1,  const float* __restrict__ qb2,
                          float* __restrict__ bias)
{
  int i = blockIdx.x * blockDim.x + threadIdx.x;
  if (i >= BIAS_TOTAL) return;
  float v = 0.f;
  if (i < BO_IN)      { int g = i/208, idx = i%208; if (idx < 200) v = b_ih[g*200+idx] + b_hh[g*200+idx]; }
  else if (i < BO_HN) { int idx = i - BO_IN; if (idx < 200) v = b_ih[400 + idx]; }
  else if (i < BO_P1) { int idx = i - BO_HN; if (idx < 200) v = b_hh[400 + idx]; }
  else if (i < BO_P2) { int idx = i - BO_P1; if (idx < 200) v = pb1[idx]; }
  else if (i < BO_Q1) { int idx = i - BO_P2; int g = idx/32, k = idx%32; if (k < 30) v = pb2[g*30 + k]; }
  else if (i < BO_Q2) { int idx = i - BO_Q1; if (idx < 200) v = qb1[idx]; }
  else                { int idx = i - BO_Q2; int g = idx/32, k = idx%32; if (k < 30) v = qb2[g*30 + k]; }
  bias[i] = v;
}

// =====================================================================
// Fragment loaders + math helpers
// =====================================================================
__device__ __forceinline__ v16h loadB(const _Float16* __restrict__ frags, int fragIdx, int lane) {
  const _Float16* p = frags + (size_t)fragIdx * 512 + lane * 16;
  v8h lo = *(const v8h*)p;
  v8h hi = *(const v8h*)(p + 8);
  v16h b;
#pragma unroll
  for (int i = 0; i < 8; ++i) { b[i] = lo[i]; b[8 + i] = hi[i]; }
  return b;
}

// A-fragment (16-bit 16xK=32): lane l row m=l%16; halves 0..7 -> k=base..base+7,
// halves 8..15 -> k=16+base..16+base+7, base=(l/16)*8.  buf is [16][Kp] f16 in LDS.
__device__ __forceinline__ v16h loadA(const _Float16* buf, int Kp, int kt, int lane) {
  int m = lane & 15;
  int base = kt*32 + (lane >> 4)*8;
  v8h lo = *(const v8h*)(buf + m*Kp + base);
  v8h hi = *(const v8h*)(buf + m*Kp + base + 16);
  v16h a;
#pragma unroll
  for (int i = 0; i < 8; ++i) { a[i] = lo[i]; a[8 + i] = hi[i]; }
  return a;
}

__device__ __forceinline__ v8f wmma(v16h a, v16h b, v8f c) {
  return __builtin_amdgcn_wmma_f32_16x16x32_f16(false, a, false, b, (short)0, c, false, false);
}

__device__ __forceinline__ float fsig(float x)  { return 1.f / (1.f + __expf(-x)); }
__device__ __forceinline__ float ftanh(float x) {
  float e = __expf(-2.f * fabsf(x));
  float r = (1.f - e) / (1.f + e);
  return x < 0.f ? -r : r;
}
__device__ __forceinline__ float fstd(float pre) {
  float sp = pre > 15.f ? pre : __logf(1.f + __expf(pre));
  return fmaxf(sp + 1e-5f, 1e-4f);
}

// =====================================================================
// Main RSSM kernel: 1 wave per block; wave owns batch rows [b0, b0+16).
// All LDS is wave-private; sequential loop over t with no inter-block sync.
// =====================================================================
__global__ void __launch_bounds__(32) rssm_kernel(
    const float* __restrict__ obs, const float* __restrict__ act,
    const float* __restrict__ noise,
    const _Float16* __restrict__ frags, const float* __restrict__ bias,
    float* __restrict__ out)
{
  extern __shared__ _Float16 smem[];
  _Float16* xa  = smem;            // [16][64]  : s(30) | a(6) | pad
  _Float16* hA  = xa  + 16*64;     // [16][224] : h staging (double buffered)
  _Float16* hB  = hA  + 16*224;
  _Float16* oa  = hB  + 16*224;    // [16][224] : obs staging
  _Float16* hid = oa  + 16*224;    // [16][224] : MLP hidden staging

  const int lane  = threadIdx.x;
  const int b0    = blockIdx.x * 16;
  const int col   = lane & 15;
  const int mbase = (lane >> 4) * 8;

  // zero all buffers (pad columns stay zero forever; h0 = s0 = 0)
  for (int i = lane; i < 16*64 + 4*16*224; i += 32) smem[i] = (_Float16)0.f;
  __syncthreads();

  _Float16* hOld = hA;
  _Float16* hNew = hB;

#pragma unroll 1
  for (int t = 0; t < L_; ++t) {
    // ---- stage actions into xa[:,30:36] ----
    for (int i = lane; i < 16*ACT_; i += 32) {
      int m = i / ACT_, c = i % ACT_;
      xa[m*64 + STOCH + c] = (_Float16)act[((size_t)(b0 + m)*L_ + t)*ACT_ + c];
    }
    // ---- stage obs into oa[:,0:200] (float4 loads) ----
    for (int i = lane; i < 16*(OBS_/4); i += 32) {
      int m = i / (OBS_/4), c4 = i % (OBS_/4);
      float4 v = ((const float4*)(obs + ((size_t)(b0 + m)*L_ + t)*OBS_))[c4];
      oa[m*224 + c4*4 + 0] = (_Float16)v.x;
      oa[m*224 + c4*4 + 1] = (_Float16)v.y;
      oa[m*224 + c4*4 + 2] = (_Float16)v.z;
      oa[m*224 + c4*4 + 3] = (_Float16)v.w;
    }
    __syncthreads();

    // ================= GRU: per 16-wide column tile =================
#pragma unroll 1
    for (int jx = 0; jx < 13; ++jx) {
      v8f aR = {}, aZ = {}, aI = {}, aH = {};
#pragma unroll
      for (int kt = 0; kt < 2; ++kt) {              // x part (K zone 0..64)
        v16h a = loadA(xa, 64, kt, lane);
        aR = wmma(a, loadB(frags, F_WRZ + jx*9        + kt, lane), aR);
        aZ = wmma(a, loadB(frags, F_WRZ + (13 + jx)*9 + kt, lane), aZ);
        aI = wmma(a, loadB(frags, F_WIN + jx*2        + kt, lane), aI);
      }
#pragma unroll 1
      for (int kt = 0; kt < 7; ++kt) {              // h part (K zone 64..288)
        v16h a = loadA(hOld, 224, kt, lane);
        aR = wmma(a, loadB(frags, F_WRZ + jx*9        + 2 + kt, lane), aR);
        aZ = wmma(a, loadB(frags, F_WRZ + (13 + jx)*9 + 2 + kt, lane), aZ);
        aH = wmma(a, loadB(frags, F_WHN + jx*7        + kt,     lane), aH);
      }
      int n = jx*16 + col;
      float br = bias[BO_RZ + n], bz = bias[BO_RZ + 208 + n];
      float bi = bias[BO_IN + n], bh = bias[BO_HN + n];
#pragma unroll
      for (int r = 0; r < 8; ++r) {
        int m = mbase + r;
        float rg = fsig(aR[r] + br);
        float zg = fsig(aZ[r] + bz);
        float ng = ftanh(aI[r] + bi + rg*(aH[r] + bh));
        float ho = (float)hOld[m*224 + n];
        float hv = (1.f - zg)*ng + zg*ho;
        hNew[m*224 + n] = (_Float16)hv;
        if (n < DETER)
          out[((size_t)(b0 + m)*L_ + t)*OUTC + 150 + n] = hv;
      }
    }
    __syncthreads();

    // ================= prior hidden: relu(h_new @ pW1^T + pb1) =================
#pragma unroll 1
    for (int jx = 0; jx < 13; ++jx) {
      v8f acc = {};
#pragma unroll 1
      for (int kt = 0; kt < 7; ++kt)
        acc = wmma(loadA(hNew, 224, kt, lane), loadB(frags, F_PW1 + jx*7 + kt, lane), acc);
      int n = jx*16 + col;
      float bb = bias[BO_P1 + n];
#pragma unroll
      for (int r = 0; r < 8; ++r) {
        float v = acc[r] + bb;
        hid[(mbase + r)*224 + n] = (_Float16)(v > 0.f ? v : 0.f);
      }
    }
    __syncthreads();

    // ================= prior out: mu_p, std_p =================
    {
      v8f P0 = {}, P1 = {}, P2 = {}, P3 = {};
#pragma unroll 1
      for (int kt = 0; kt < 7; ++kt) {
        v16h a = loadA(hid, 224, kt, lane);
        P0 = wmma(a, loadB(frags, F_PW2 + 0*7 + kt, lane), P0);
        P1 = wmma(a, loadB(frags, F_PW2 + 1*7 + kt, lane), P1);
        P2 = wmma(a, loadB(frags, F_PW2 + 2*7 + kt, lane), P2);
        P3 = wmma(a, loadB(frags, F_PW2 + 3*7 + kt, lane), P3);
      }
#pragma unroll
      for (int r = 0; r < 8; ++r) {
        int m = mbase + r;
        size_t ofs = ((size_t)(b0 + m)*L_ + t)*OUTC;
        // jj=0: n=col (tiles P0/P2); jj=1: n=16+col (tiles P1/P3)
        {
          int n = col;
          float mu  = P0[r] + bias[BO_P2 + n];
          float std = fstd(P2[r] + bias[BO_P2 + 32 + n]);
          if (n < STOCH) { out[ofs + n] = mu; out[ofs + 30 + n] = std; }
        }
        {
          int n = 16 + col;
          float mu  = P1[r] + bias[BO_P2 + n];
          float std = fstd(P3[r] + bias[BO_P2 + 32 + n]);
          if (n < STOCH) { out[ofs + n] = mu; out[ofs + 30 + n] = std; }
        }
      }
    }
    __syncthreads();

    // ================= posterior hidden: relu([obs,h_new] @ qW1^T + qb1) =================
#pragma unroll 1
    for (int jx = 0; jx < 13; ++jx) {
      v8f acc = {};
#pragma unroll 1
      for (int kt = 0; kt < 7; ++kt)
        acc = wmma(loadA(oa, 224, kt, lane),   loadB(frags, F_QW1 + jx*14 + kt,     lane), acc);
#pragma unroll 1
      for (int kt = 0; kt < 7; ++kt)
        acc = wmma(loadA(hNew, 224, kt, lane), loadB(frags, F_QW1 + jx*14 + 7 + kt, lane), acc);
      int n = jx*16 + col;
      float bb = bias[BO_Q1 + n];
#pragma unroll
      for (int r = 0; r < 8; ++r) {
        float v = acc[r] + bb;
        hid[(mbase + r)*224 + n] = (_Float16)(v > 0.f ? v : 0.f);
      }
    }
    __syncthreads();

    // ================= posterior out: mu_q, std_q, s_new =================
    {
      v8f Q0 = {}, Q1 = {}, Q2 = {}, Q3 = {};
#pragma unroll 1
      for (int kt = 0; kt < 7; ++kt) {
        v16h a = loadA(hid, 224, kt, lane);
        Q0 = wmma(a, loadB(frags, F_QW2 + 0*7 + kt, lane), Q0);
        Q1 = wmma(a, loadB(frags, F_QW2 + 1*7 + kt, lane), Q1);
        Q2 = wmma(a, loadB(frags, F_QW2 + 2*7 + kt, lane), Q2);
        Q3 = wmma(a, loadB(frags, F_QW2 + 3*7 + kt, lane), Q3);
      }
#pragma unroll
      for (int r = 0; r < 8; ++r) {
        int m = mbase + r;
        size_t ofs = ((size_t)(b0 + m)*L_ + t)*OUTC;
        {
          int n = col;
          float mu  = Q0[r] + bias[BO_Q2 + n];
          float std = fstd(Q2[r] + bias[BO_Q2 + 32 + n]);
          if (n < STOCH) {
            float eps = noise[((size_t)(b0 + m)*L_ + t)*STOCH + n];
            float s = mu + std*eps;
            out[ofs + 60 + n] = mu; out[ofs + 90 + n] = std; out[ofs + 120 + n] = s;
            xa[m*64 + n] = (_Float16)s;                 // stage s for next step
          }
        }
        {
          int n = 16 + col;
          float mu  = Q1[r] + bias[BO_Q2 + n];
          float std = fstd(Q3[r] + bias[BO_Q2 + 32 + n]);
          if (n < STOCH) {
            float eps = noise[((size_t)(b0 + m)*L_ + t)*STOCH + n];
            float s = mu + std*eps;
            out[ofs + 60 + n] = mu; out[ofs + 90 + n] = std; out[ofs + 120 + n] = s;
            xa[m*64 + n] = (_Float16)s;
          }
        }
      }
    }
    __syncthreads();

    // swap h double buffers
    _Float16* tmp = hOld; hOld = hNew; hNew = tmp;
  }
}

// =====================================================================
extern "C" void kernel_launch(void* const* d_in, const int* in_sizes, int n_in,
                              void* d_out, int out_size, void* d_ws, size_t ws_size,
                              hipStream_t stream) {
  const float* obs   = (const float*)d_in[0];
  const float* act   = (const float*)d_in[1];
  const float* noise = (const float*)d_in[2];
  const float* W_ih  = (const float*)d_in[3];
  const float* b_ih  = (const float*)d_in[4];
  const float* W_hh  = (const float*)d_in[5];
  const float* b_hh  = (const float*)d_in[6];
  const float* pW1   = (const float*)d_in[7];
  const float* pb1   = (const float*)d_in[8];
  const float* pW2   = (const float*)d_in[9];
  const float* pb2   = (const float*)d_in[10];
  const float* qW1   = (const float*)d_in[11];
  const float* qb1   = (const float*)d_in[12];
  const float* qW2   = (const float*)d_in[13];
  const float* qb2   = (const float*)d_in[14];
  // d_in[15], d_in[16] (sW, sb) unused by the reference scan.

  _Float16* frags = (_Float16*)d_ws;
  float*    bias  = (float*)((char*)d_ws + FRAG_BYTES);

  pack_frags<<<F_TOTAL, 64, 0, stream>>>(W_ih, W_hh, pW1, pW2, qW1, qW2, frags);
  pack_bias<<<(BIAS_TOTAL + 255) / 256, 256, 0, stream>>>(b_ih, b_hh, pb1, pb2, qb1, qb2, bias);

  const int ldsBytes = (16*64 + 4*16*224) * (int)sizeof(_Float16); // 30720 B per wave-block
  rssm_kernel<<<B_/16, 32, ldsBytes, stream>>>(obs, act, noise, frags, bias, (float*)d_out);
}
